// sgdnetIDT_644245094840
// MI455X (gfx1250) — matrix operand order repaired and checked by reading, not compiled, not run on hardware
//
#include <hip/hip_runtime.h>
#include <hip/hip_bf16.h>

typedef __attribute__((ext_vector_type(16))) _Float16 v16h;
typedef __attribute__((ext_vector_type(8)))  float    v8f;
typedef __attribute__((ext_vector_type(4)))  unsigned int u32x4;
typedef __attribute__((ext_vector_type(8)))  int      v8i;
typedef __attribute__((ext_vector_type(4)))  int      v4i;

#define BB 4
#define HH 256
#define WW 256
#define HW 65536            // H*W
#define NBF 40
#define DPB 131072          // floats per batch in flat view (C*H*W*2)
#define SLOT 524288         // floats per slot (B*DPB)

#if __has_builtin(__builtin_amdgcn_tensor_load_to_lds) && __has_builtin(__builtin_amdgcn_s_wait_tensorcnt)
#define USE_TDM 1
#else
#define USE_TDM 0
#endif

// ---------------------------------------------------------------------------
// 256-point radix-2 DIT FFT in LDS, one row/col per 128-thread block (4 waves)
// sign = +1 for inverse (e^{+i}), -1 for forward. scale applied at store.
// ---------------------------------------------------------------------------
__device__ inline void fft256_stages(float2* s, int tid, float sign) {
#pragma unroll
  for (int st = 1; st <= 8; ++st) {
    __syncthreads();
    int m = 1 << st, half = m >> 1;
    int grp = tid >> (st - 1);
    int k = tid & (half - 1);
    int i1 = grp * m + k;
    int i2 = i1 + half;
    float a = sign * 6.28318530717958647692f * (float)k / (float)m;
    float wi, wr;
    sincosf(a, &wi, &wr);
    float2 x1 = s[i1], x2 = s[i2];
    float tr = wr * x2.x - wi * x2.y;
    float ti = wr * x2.y + wi * x2.x;
    s[i1] = make_float2(x1.x + tr, x1.y + ti);
    s[i2] = make_float2(x1.x - tr, x1.y - ti);
  }
  __syncthreads();
}

__global__ __launch_bounds__(128)
void fft1d_row(const float2* __restrict__ srcC, const float* __restrict__ srcR,
               float2* __restrict__ dst, float sign, float scale) {
  __shared__ float2 s[256];
  int row = blockIdx.x;            // b*256 + h
  int base = row << 8;
  int tid = threadIdx.x;
#pragma unroll
  for (int q = 0; q < 2; ++q) {
    int i = tid + (q << 7);
    int r = __brev((unsigned)i) >> 24;
    float2 v;
    if (srcR) v = make_float2(srcR[base + i], 0.f);
    else      v = srcC[base + i];
    s[r] = v;
  }
  fft256_stages(s, tid, sign);
#pragma unroll
  for (int q = 0; q < 2; ++q) {
    int i = tid + (q << 7);
    float2 v = s[i];
    dst[base + i] = make_float2(v.x * scale, v.y * scale);
  }
}

// Column pass: 2KB-strided gather. On CDNA5 we hand the strided gather to the
// Tensor Data Mover: one 2D descriptor (8B elements, tile 1x256, row stride
// 256 elems) DMAs the column into LDS; wave 0 issues, waits TENSORcnt, then
// the block barriers and bit-reverses locally.
__global__ __launch_bounds__(128)
void fft1d_col(const float2* __restrict__ src, float2* __restrict__ dst,
               float sign, float scale) {
  __shared__ float2 s[256];
  int col = blockIdx.x;            // b*256 + w
  int b = col >> 8, w = col & 255;
  int base = (b << 16) + w;
  int tid = threadIdx.x;
#if USE_TDM
  __shared__ float2 sraw[256];
  if (tid < 32) {                  // wave 0 issues the tensor DMA
    unsigned lds_addr = (unsigned)(size_t)(&sraw[0]);
    unsigned long long ga = (unsigned long long)(size_t)(src + base);
    u32x4 g0;
    g0[0] = 1u;                                              // count=1, user D#
    g0[1] = lds_addr;                                        // lds_addr [63:32]
    g0[2] = (unsigned)ga;                                    // global_addr lo
    g0[3] = (unsigned)((ga >> 32) & 0x01FFFFFFu) | (2u << 30); // ga hi | type=2
    v8i g1;
    g1[0] = (int)(3u << 16);        // wg_mask=0, data_size=3 (8B)
    g1[1] = (int)(256u << 16);      // tensor_dim0[15:0]=256 in [31:16]
    g1[2] = (int)(256u << 16);      // tensor_dim0 hi=0 | tensor_dim1 lo=256
    g1[3] = (int)(1u << 16);        // tensor_dim1 hi=0 | tile_dim0=1
    g1[4] = 256;                    // tile_dim1=256, tile_dim2=0
    g1[5] = 256;                    // tensor_dim0_stride lo = 256 (elements)
    g1[6] = 0;                      // stride0 hi | tensor_dim1_stride lo
    g1[7] = 0;
    v4i gz4 = {0, 0, 0, 0};
    v8i gz8 = {0, 0, 0, 0, 0, 0, 0, 0};
    __builtin_amdgcn_tensor_load_to_lds(g0, g1, gz4, gz4, gz8, 0);
    __builtin_amdgcn_s_wait_tensorcnt(0);
  }
  __syncthreads();
#pragma unroll
  for (int q = 0; q < 2; ++q) {
    int i = tid + (q << 7);
    s[__brev((unsigned)i) >> 24] = sraw[i];
  }
#else
#pragma unroll
  for (int q = 0; q < 2; ++q) {
    int i = tid + (q << 7);
    s[__brev((unsigned)i) >> 24] = src[base + (i << 8)];
  }
#endif
  fft256_stages(s, tid, sign);
#pragma unroll
  for (int q = 0; q < 2; ++q) {
    int i = tid + (q << 7);
    float2 v = s[i];
    dst[base + (i << 8)] = make_float2(v.x * scale, v.y * scale);
  }
}

// ---------------------------------------------------------------------------
// Per-batch normalize of real part: zr = (re - min) / (max - min)
// ---------------------------------------------------------------------------
__global__ __launch_bounds__(256)
void normalize_k(const float2* __restrict__ src, float* __restrict__ zr,
                 float* __restrict__ periter) {
  __shared__ float smn[256], smx[256];
  int b = blockIdx.x;
  int tid = threadIdx.x;
  int base = b << 16;
  float mn = 3.0e38f, mx = -3.0e38f;
  for (int i = tid; i < HW; i += 256) {
    float v = src[base + i].x;
    mn = fminf(mn, v);
    mx = fmaxf(mx, v);
  }
  smn[tid] = mn; smx[tid] = mx;
  __syncthreads();
  for (int o = 128; o; o >>= 1) {
    if (tid < o) {
      smn[tid] = fminf(smn[tid], smn[tid + o]);
      smx[tid] = fmaxf(smx[tid], smx[tid + o]);
    }
    __syncthreads();
  }
  mn = smn[0]; mx = smx[0];
  float inv = 1.f / (mx - mn);
  for (int i = tid; i < HW; i += 256) {
    float v = (src[base + i].x - mn) * inv;
    zr[base + i] = v;
    if (periter && b == 0) periter[i] = v;
  }
}

// ---------------------------------------------------------------------------
// Pack conv weights into WMMA A-fragment layouts (f16), per-lane contiguous.
// ---------------------------------------------------------------------------
__device__ inline int a_kmap(int g, int e) {
  int v = e >> 1, p = e & 1;
  return g * 16 + v * 2 + p;
}

__global__ void pack_weights(const float* __restrict__ W1, const float* __restrict__ W2,
                             _Float16* __restrict__ fA1, _Float16* __restrict__ fA2) {
  int t = threadIdx.x;
  for (int idx = t; idx < 2 * 32 * 16; idx += blockDim.x) {
    int e = idx & 15, L = (idx >> 4) & 31, f = idx >> 9;
    int m = L & 15, g = L >> 4;
    int K = a_kmap(g, e);
    float val = 0.f;
    if (K < 9) val = W1[(f * 16 + m) * 9 + K];
    fA1[idx] = (_Float16)val;
  }
  for (int idx = t; idx < 9 * 32 * 16; idx += blockDim.x) {
    int e = idx & 15, L = (idx >> 4) & 31, c = idx >> 9;
    int m = L & 15, g = L >> 4;
    int K = a_kmap(g, e);
    int kg = c * 32 + K;                      // flat k = ci*9 + tap
    float val = 0.f;
    if (m == 0 && kg < 288) val = W2[kg];
    fA2[idx] = (_Float16)val;
  }
}

// ---------------------------------------------------------------------------
// conv1: 1 -> 32 channels, 3x3 SAME, bias + ReLU, implicit GEMM via WMMA.
// 128-thread block = 4 waves, one 16-pixel tile per wave.
// B 32x16 f16 layout: lane -> col N=L&15, K = (L>>4)*16 + elem.
// ---------------------------------------------------------------------------
__global__ __launch_bounds__(128)
void conv1_wmma(const float* __restrict__ zr, const _Float16* __restrict__ fA1,
                const float* __restrict__ b1, _Float16* __restrict__ hidden) {
  int tile = blockIdx.x * 4 + (threadIdx.x >> 5);
  int lane = threadIdx.x & 31;
  int w0 = (tile & 15) << 4;
  int h  = (tile >> 4) & 255;
  int b  = tile >> 12;
  const float* zb = zr + ((size_t)b << 16);
  int n = lane & 15, g = lane >> 4;
  v16h bf;
#pragma unroll
  for (int e = 0; e < 16; ++e) {
    int K = (g << 4) + e;
    float val = 0.f;
    if (K < 9) {
      int dy = K / 3 - 1, dx = K % 3 - 1;
      int hh = h + dy, ww = w0 + n + dx;
      if (hh >= 0 && hh < HH && ww >= 0 && ww < WW) val = zb[(hh << 8) + ww];
    }
    bf[e] = (_Float16)val;
  }
  v16h a0 = *(const v16h*)(fA1 + (size_t)lane * 16);
  v16h a1 = *(const v16h*)(fA1 + (size_t)(32 + lane) * 16);
  v8f c0 = {}, c1 = {};
  c0 = __builtin_amdgcn_wmma_f32_16x16x32_f16(false, a0, false, bf, (short)0, c0, false, false);
  c1 = __builtin_amdgcn_wmma_f32_16x16x32_f16(false, a1, false, bf, (short)0, c1, false, false);
  int pix = (h << 8) + w0 + n;
#pragma unroll
  for (int r = 0; r < 8; ++r) {
    int mrow = r + (g << 3);
    float v0 = fmaxf(c0[r] + b1[mrow], 0.f);
    float v1 = fmaxf(c1[r] + b1[16 + mrow], 0.f);
    hidden[(((size_t)b * 32 + mrow) << 16) + pix] = (_Float16)v0;
    hidden[(((size_t)b * 32 + 16 + mrow) << 16) + pix] = (_Float16)v1;
  }
}

// ---------------------------------------------------------------------------
// conv2: 32 -> 1 channel, 3x3 SAME + bias. K = 288 padded to 9 chunks of 32.
// ---------------------------------------------------------------------------
__global__ __launch_bounds__(128)
void conv2_wmma(const _Float16* __restrict__ hidden, const _Float16* __restrict__ fA2,
                const float* __restrict__ b2, float* __restrict__ outp) {
  int tile = blockIdx.x * 4 + (threadIdx.x >> 5);
  int lane = threadIdx.x & 31;
  int w0 = (tile & 15) << 4;
  int h  = (tile >> 4) & 255;
  int b  = tile >> 12;
  int n = lane & 15, g = lane >> 4;
  v8f acc = {};
#pragma unroll
  for (int c = 0; c < 9; ++c) {
    v16h a = *(const v16h*)(fA2 + (size_t)(c * 32 + lane) * 16);
    v16h bf;
#pragma unroll
    for (int e = 0; e < 16; ++e) {
      int kg = c * 32 + (g << 4) + e;
      _Float16 val = (_Float16)0.f;
      if (kg < 288) {
        int ci = kg / 9, t = kg % 9;
        int dy = t / 3 - 1, dx = t % 3 - 1;
        int hh = h + dy, ww = w0 + n + dx;
        if (hh >= 0 && hh < HH && ww >= 0 && ww < WW)
          val = hidden[(((size_t)b * 32 + ci) << 16) + (hh << 8) + ww];
      }
      bf[e] = val;
    }
    acc = __builtin_amdgcn_wmma_f32_16x16x32_f16(false, a, false, bf, (short)0, acc, false, false);
  }
  if (lane < 16)
    outp[((size_t)b << 16) + (h << 8) + w0 + n] = acc[0] + b2[0];
}

// ---------------------------------------------------------------------------
// Fused data term + RED update (dominant HBM stream -> software prefetch):
// znew = z - gamma*( mean_k conj(H)*(H*z - y) + tau*(z - den) )
// ---------------------------------------------------------------------------
__global__ __launch_bounds__(256)
void ured_update(const float2* __restrict__ z, const float2* __restrict__ den,
                 const float2* __restrict__ y, const float2* __restrict__ Hc,
                 const float* __restrict__ gamma_p, const float* __restrict__ tau_p,
                 float2* __restrict__ out) {
  int idx = blockIdx.x * blockDim.x + threadIdx.x;
  if (idx >= BB * HW) return;
  int b = idx >> 16, x = idx & 65535;
  float2 zv = z[idx];
  float ar = 0.f, ai = 0.f;
  const float2* yb = y + ((size_t)b * NBF) * HW;
  for (int k = 0; k < NBF; ++k) {
    if (k + 4 < NBF) {
      __builtin_prefetch(&Hc[(size_t)(k + 4) * HW + x], 0, 0);
      __builtin_prefetch(&yb[(size_t)(k + 4) * HW + x], 0, 0);
    }
    float2 hv = Hc[(size_t)k * HW + x];
    float2 yv = yb[(size_t)k * HW + x];
    float hzr = hv.x * zv.x - hv.y * zv.y;
    float hzi = hv.x * zv.y + hv.y * zv.x;
    float dr = hzr - yv.x, di = hzi - yv.y;
    ar += hv.x * dr + hv.y * di;          // conj(h) * d
    ai += hv.x * di - hv.y * dr;
  }
  float gg = gamma_p[0], tt = tau_p[0];
  ar *= (1.f / (float)NBF);
  ai *= (1.f / (float)NBF);
  float2 dv = den[idx];
  float ur = ar + tt * (zv.x - dv.x);
  float ui = ai + tt * (zv.y - dv.y);
  out[idx] = make_float2(zv.x - gg * ur, zv.y - gg * ui);
}

// ---------------------------------------------------------------------------
// Anderson: Gram matrix GG[b,i,j] = <F_i - X_i, F_j - X_j>
// ---------------------------------------------------------------------------
__global__ __launch_bounds__(256)
void gram_k(const float* __restrict__ Xs, const float* __restrict__ Fs,
            int n, float* __restrict__ GG) {
  __shared__ float red[256];
  int bi = blockIdx.x;
  int j = bi % n;
  int i = (bi / n) % n;
  int b = bi / (n * n);
  size_t off = (size_t)b * DPB;
  const float* Xi = Xs + (size_t)i * SLOT + off;
  const float* Fi = Fs + (size_t)i * SLOT + off;
  const float* Xj = Xs + (size_t)j * SLOT + off;
  const float* Fj = Fs + (size_t)j * SLOT + off;
  float s = 0.f;
  for (int e = threadIdx.x; e < DPB; e += 256)
    s += (Fi[e] - Xi[e]) * (Fj[e] - Xj[e]);
  red[threadIdx.x] = s;
  __syncthreads();
  for (int o = 128; o; o >>= 1) {
    if (threadIdx.x < o) red[threadIdx.x] += red[threadIdx.x + o];
    __syncthreads();
  }
  if (threadIdx.x == 0) GG[b * 25 + i * 5 + j] = red[0];
}

// Tiny per-batch (n+1)x(n+1) solve with partial pivoting -> alpha[b,0..n-1]
__global__ void solve_k(const float* __restrict__ GG, int n, float* __restrict__ alpha) {
  int b = threadIdx.x;
  if (b >= BB) return;
  float Hm[6][7];
  int N = n + 1;
  for (int r = 0; r < N; ++r)
    for (int c = 0; c <= N; ++c) Hm[r][c] = 0.f;
  for (int c = 1; c < N; ++c) Hm[0][c] = 1.f;
  for (int r = 1; r < N; ++r) Hm[r][0] = 1.f;
  for (int i = 0; i < n; ++i)
    for (int j = 0; j < n; ++j)
      Hm[i + 1][j + 1] = GG[b * 25 + i * 5 + j] + (i == j ? 1.0e-4f : 0.f);
  Hm[0][N] = 1.f;
  for (int col = 0; col < N; ++col) {
    int piv = col;
    for (int r = col + 1; r < N; ++r)
      if (fabsf(Hm[r][col]) > fabsf(Hm[piv][col])) piv = r;
    if (piv != col)
      for (int c = col; c <= N; ++c) { float t = Hm[col][c]; Hm[col][c] = Hm[piv][c]; Hm[piv][c] = t; }
    float inv = 1.f / Hm[col][col];
    for (int c = col; c <= N; ++c) Hm[col][c] *= inv;
    for (int r = 0; r < N; ++r) {
      if (r == col) continue;
      float f = Hm[r][col];
      if (f != 0.f)
        for (int c = col; c <= N; ++c) Hm[r][c] -= f * Hm[col][c];
    }
  }
  for (int i = 0; i < n; ++i) alpha[b * 5 + i] = Hm[i + 1][N];
}

// xnew = sum_i alpha_i * (beta*F_i + (1-beta)*X_i)
__global__ __launch_bounds__(256)
void combine_k(const float* __restrict__ Xs, const float* __restrict__ Fs,
               const float* __restrict__ alpha, int n, float beta,
               float* __restrict__ xnew) {
  int idx = blockIdx.x * 256 + threadIdx.x;
  if (idx >= SLOT) return;
  int b = idx >> 17;
  size_t off = (size_t)idx;
  float s = 0.f;
  for (int i = 0; i < n; ++i) {
    float a = alpha[b * 5 + i];
    s += a * (beta * Fs[(size_t)i * SLOT + off] + (1.f - beta) * Xs[(size_t)i * SLOT + off]);
  }
  xnew[off] = s;
}

__global__ __launch_bounds__(256)
void writeout_k(const float2* __restrict__ src, float* __restrict__ out) {
  int idx = blockIdx.x * 256 + threadIdx.x;
  if (idx >= BB * HW) return;
  float2 v = src[idx];
  out[idx * 2] = v.x;
  out[idx * 2 + 1] = 0.f;
}

// ---------------------------------------------------------------------------
extern "C" void kernel_launch(void* const* d_in, const int* in_sizes, int n_in,
                              void* d_out, int out_size, void* d_ws, size_t ws_size,
                              hipStream_t stream) {
  const float2* n_ipt = (const float2*)d_in[0];
  const float2* n_y   = (const float2*)d_in[1];
  const float2* Hreal = (const float2*)d_in[2];
  const float*  W1    = (const float*)d_in[3];
  const float*  b1    = (const float*)d_in[4];
  const float*  W2    = (const float*)d_in[5];
  const float*  b2    = (const float*)d_in[6];
  const float*  gamma = (const float*)d_in[7];
  const float*  tau   = (const float*)d_in[8];
  float* out = (float*)d_out;

  size_t off = 0;
  auto wsalloc = [&](size_t bytes) -> void* {
    void* p = (void*)((char*)d_ws + off);
    off += (bytes + 255) & ~(size_t)255;
    return p;
  };
  float2* t1      = (float2*)wsalloc((size_t)SLOT * 4);
  float2* t2      = (float2*)wsalloc((size_t)SLOT * 4);
  float*  Xs      = (float*) wsalloc((size_t)5 * SLOT * 4);
  float*  Fs      = (float*) wsalloc((size_t)5 * SLOT * 4);
  float*  zr      = (float*) wsalloc((size_t)BB * HW * 4);
  float*  cnnout  = (float*) wsalloc((size_t)BB * HW * 4);
  float2* den     = (float2*)wsalloc((size_t)SLOT * 4);
  float*  GG      = (float*) wsalloc(100 * 4);
  float*  alphad  = (float*) wsalloc(20 * 4);
  _Float16* hidden = (_Float16*)wsalloc((size_t)BB * 32 * HW * 2);
  _Float16* fA1    = (_Float16*)wsalloc(1024 * 2);
  _Float16* fA2    = (_Float16*)wsalloc(4608 * 2);

  const int FROWS = BB * HH;          // 1024 row/col FFT blocks
  const int TILEB = BB * HH * (WW / 16) / 4;   // 4 tiles (waves) per block

  auto f_eval = [&](const float* xin, float* fout) {
    const float2* zc = (const float2*)xin;
    // zr = normalize(real(ifft2(z)))
    fft1d_row<<<FROWS, 128, 0, stream>>>(zc, (const float*)nullptr, t1, +1.f, 1.f / 256.f);
    fft1d_col<<<FROWS, 128, 0, stream>>>(t1, t2, +1.f, 1.f / 256.f);
    normalize_k<<<BB, 256, 0, stream>>>(t2, zr, (float*)nullptr);
    // cnn
    conv1_wmma<<<TILEB, 128, 0, stream>>>(zr, fA1, b1, hidden);
    conv2_wmma<<<TILEB, 128, 0, stream>>>(hidden, fA2, b2, cnnout);
    // den = fft2(cnn)
    fft1d_row<<<FROWS, 128, 0, stream>>>((const float2*)nullptr, cnnout, t1, -1.f, 1.f);
    fft1d_col<<<FROWS, 128, 0, stream>>>(t1, den, -1.f, 1.f);
    // z - gamma*(delta_g + tau*(z - den))
    ured_update<<<(BB * HW) / 256, 256, 0, stream>>>(zc, den, n_y, Hreal, gamma, tau,
                                                     (float2*)fout);
  };

  // ---- setup ----
  pack_weights<<<1, 256, 0, stream>>>(W1, W2, fA1, fA2);
  // nr = normalize(real(ifft2(n_ipt))); n_ipt_periter = nr[0]
  fft1d_row<<<FROWS, 128, 0, stream>>>(n_ipt, (const float*)nullptr, t1, +1.f, 1.f / 256.f);
  fft1d_col<<<FROWS, 128, 0, stream>>>(t1, t2, +1.f, 1.f / 256.f);
  normalize_k<<<BB, 256, 0, stream>>>(t2, zr, out + (size_t)BB * HW * 2);
  // z0 = fft2(nr) -> Xs[0]
  fft1d_row<<<FROWS, 128, 0, stream>>>((const float2*)nullptr, zr, t1, -1.f, 1.f);
  fft1d_col<<<FROWS, 128, 0, stream>>>(t1, (float2*)Xs, -1.f, 1.f);

  // ---- Anderson acceleration ----
  f_eval(Xs, Fs);                                                   // Fs[0] = f(Xs[0])
  (void)hipMemcpyAsync(Xs + SLOT, Fs, (size_t)SLOT * 4,
                       hipMemcpyDeviceToDevice, stream);            // Xs[1] = Fs[0]
  f_eval(Xs + SLOT, Fs + SLOT);                                     // Fs[1] = f(Xs[1])
  for (int k = 2; k < 10; ++k) {
    int n = (k < 5) ? k : 5;
    int last = k % 5;
    gram_k<<<BB * n * n, 256, 0, stream>>>(Xs, Fs, n, GG);
    solve_k<<<1, BB, 0, stream>>>(GG, n, alphad);
    combine_k<<<SLOT / 256, 256, 0, stream>>>(Xs, Fs, alphad, n, 1.0f,
                                              Xs + (size_t)last * SLOT);
    f_eval(Xs + (size_t)last * SLOT, Fs + (size_t)last * SLOT);
  }
  // final z = f(Xs[4]) == Fs[4] (computed at k=9); out = real(ifft2(z))
  fft1d_row<<<FROWS, 128, 0, stream>>>((const float2*)(Fs + (size_t)4 * SLOT),
                                       (const float*)nullptr, t1, +1.f, 1.f / 256.f);
  fft1d_col<<<FROWS, 128, 0, stream>>>(t1, t2, +1.f, 1.f / 256.f);
  writeout_k<<<(BB * HW) / 256, 256, 0, stream>>>(t2, out);
}
